// SeparateSparseMixedAttn_18313740550560
// MI455X (gfx1250) — matrix operand order repaired and checked by reading, compiled, not run on hardware
//
#include <hip/hip_runtime.h>
#include <hip/hip_bf16.h>

typedef __attribute__((ext_vector_type(16))) __bf16 v16bf;
typedef __attribute__((ext_vector_type(8)))  __bf16 v8bf;
typedef __attribute__((ext_vector_type(8)))  float  v8f;

#define B_N   8
#define C_N   256
#define WH    128
#define SP    16384   // 128*128
#define CR_N  32

__device__ __forceinline__ unsigned short f2bf(float f) {
    unsigned int u = __float_as_uint(f);
    unsigned int r = u + 0x7FFFu + ((u >> 16) & 1u);   // round-to-nearest-even
    return (unsigned short)(r >> 16);
}
__device__ __forceinline__ v8bf ld8(const unsigned short* p)  { return *(const v8bf*)p; }
__device__ __forceinline__ v16bf ld16(const unsigned short* p){ return *(const v16bf*)p; }
__device__ __forceinline__ v16bf cat8(v8bf lo, v8bf hi) {
    v16bf r;
#pragma unroll
    for (int i = 0; i < 8; ++i) { r[i] = lo[i]; r[i + 8] = hi[i]; }
    return r;
}

// CDNA5 async global->LDS copy, 16B per lane, tracked by ASYNCcnt.
// Generic pointers to __shared__ carry the LDS offset in their low 32 bits
// (ISA 10.2: LDS aperture address is truncated to addr[31:0]).
__device__ __forceinline__ void async_ld_b128(const void* gptr, void* lptr) {
    unsigned lds_off = (unsigned)(uintptr_t)lptr;
    asm volatile("global_load_async_to_lds_b128 %0, %1, off"
                 :: "v"(lds_off), "v"(gptr) : "memory");
}
__device__ __forceinline__ void wait_async0() {
    asm volatile("s_wait_asynccnt 0" ::: "memory");
}

// ---------------- kernel 0: zero pool + stats (ws not re-poisoned between replays)
__global__ void init_stats_kernel(float* pool, float* stats) {
    int tid = blockIdx.x * blockDim.x + threadIdx.x;
    if (tid < B_N * C_N) pool[tid] = 0.f;
    if (tid < 2 * B_N * CR_N) stats[tid] = 0.f;
}

// ---------------- kernel 1: depthwise dilated convs (q dil=3, k dil=6) + pool partial sums
__global__ void conv_qk_kernel(const float* __restrict__ x,
                               const float* __restrict__ wq, const float* __restrict__ bq,
                               const float* __restrict__ wk, const float* __restrict__ bk,
                               unsigned short* __restrict__ qg,
                               unsigned short* __restrict__ ktg,
                               float* __restrict__ pool) {
    __shared__ float xt[44 * WH];    // 32 rows + 6 halo each side
    __shared__ float red[256];
    int blk  = blockIdx.x;
    int bc   = blk >> 2;
    int r0   = (blk & 3) * 32;
    int tid  = threadIdx.x;
    const float* xp = x + (size_t)bc * SP;
    for (int i = tid; i < 44 * WH; i += 256) {
        int lr = i >> 7, col = i & 127;
        int gr = r0 - 6 + lr;
        xt[i] = (gr >= 0 && gr < WH) ? xp[gr * WH + col] : 0.f;
    }
    __syncthreads();
    int c = bc & (C_N - 1);
    float wq9[9], wk9[9];
#pragma unroll
    for (int t = 0; t < 9; ++t) { wq9[t] = wq[c * 9 + t]; wk9[t] = wk[c * 9 + t]; }
    float bqv = bq[c], bkv = bk[c];
    float ksum = 0.f;
    for (int i = tid; i < 32 * WH; i += 256) {
        int lw = i >> 7, h = i & 127;
        int lrow = 6 + lw;
        float qv = bqv, kv = bkv;
#pragma unroll
        for (int dy = -1; dy <= 1; ++dy)
#pragma unroll
            for (int dx = -1; dx <= 1; ++dx) {
                int wi = (dy + 1) * 3 + (dx + 1);
                int h3 = h + 3 * dx;
                if (h3 >= 0 && h3 < WH) qv += xt[(lrow + 3 * dy) * WH + h3] * wq9[wi];
                int h6 = h + 6 * dx;
                if (h6 >= 0 && h6 < WH) kv += xt[(lrow + 6 * dy) * WH + h6] * wk9[wi];
            }
        int gw = r0 + lw;
        qg [(size_t)bc * SP + gw * WH + h]  = f2bf(qv);
        ktg[(size_t)bc * SP + h * WH + gw]  = f2bf(kv);   // transposed store
        ksum += kv;
    }
    red[tid] = ksum; __syncthreads();
    for (int s = 128; s > 0; s >>= 1) { if (tid < s) red[tid] += red[tid + s]; __syncthreads(); }
    if (tid == 0) atomicAdd(&pool[bc], red[0]);
}

// ---------------- kernel 2: per-(b,c) 128x128x128 GEMM (WMMA bf16) + fused plane softmax * pool
// Tile staging via async global->LDS (ASYNCcnt); 8 B-fragments prefetched per k-step so the
// 8 WMMAs issue back-to-back behind a single dscnt wait.
__global__ void qk_gemm_softmax_kernel(const unsigned short* __restrict__ qg,
                                       const unsigned short* __restrict__ ktg,
                                       const float* __restrict__ pool,
                                       unsigned short* __restrict__ yg) {
    __shared__ alignas(32) unsigned short qs[WH * 64];  // 16KB: q[:, kc*64..]
    __shared__ alignas(32) unsigned short ks[WH * 64];  // 16KB: kT[:, kc*64..]
    __shared__ float red[256];
    int bc = blockIdx.x;
    int tid = threadIdx.x;
    int wave = tid >> 5, lane = tid & 31, half = lane >> 4, l15 = lane & 15;
    const unsigned short* qp = qg  + (size_t)bc * SP;
    const unsigned short* kp = ktg + (size_t)bc * SP;
    v8f acc[8];
#pragma unroll
    for (int t = 0; t < 8; ++t)
#pragma unroll
        for (int e = 0; e < 8; ++e) acc[t][e] = 0.f;
    int M0 = wave * 16;
    for (int kc = 0; kc < 2; ++kc) {
        __syncthreads();
#pragma unroll
        for (int i = tid; i < 1024; i += 256) {       // 16B per lane per op
            int row = i >> 3, seg = i & 7;
            async_ld_b128(&qp[row * WH + kc * 64 + seg * 8], &qs[row * 64 + seg * 8]);
            async_ld_b128(&kp[row * WH + kc * 64 + seg * 8], &ks[row * 64 + seg * 8]);
        }
        wait_async0();
        __syncthreads();
#pragma unroll
        for (int kk = 0; kk < 2; ++kk) {
            int k0 = kk * 32;
            v16bf a = ld16(&qs[(M0 + l15) * 64 + k0 + half * 16]);
            v16bf bm[8];
#pragma unroll
            for (int nt = 0; nt < 8; ++nt)
                bm[nt] = ld16(&ks[(nt * 16 + l15) * 64 + k0 + half * 16]);
#pragma unroll
            for (int nt = 0; nt < 8; ++nt)
                acc[nt] = __builtin_amdgcn_wmma_f32_16x16x32_bf16(
                    false, a, false, bm[nt], (short)0, acc[nt], false, false);
        }
    }
    // block-wide softmax over the 128x128 plane held in the 8 waves' accumulators
    float lmax = -3.4e38f;
#pragma unroll
    for (int t = 0; t < 8; ++t)
#pragma unroll
        for (int e = 0; e < 8; ++e) lmax = fmaxf(lmax, acc[t][e]);
    __syncthreads();
    red[tid] = lmax; __syncthreads();
    for (int s = 128; s > 0; s >>= 1) { if (tid < s) red[tid] = fmaxf(red[tid], red[tid + s]); __syncthreads(); }
    float gmax = red[0]; __syncthreads();
    float lsum = 0.f;
#pragma unroll
    for (int t = 0; t < 8; ++t)
#pragma unroll
        for (int e = 0; e < 8; ++e) { float v = __expf(acc[t][e] - gmax); acc[t][e] = v; lsum += v; }
    red[tid] = lsum; __syncthreads();
    for (int s = 128; s > 0; s >>= 1) { if (tid < s) red[tid] += red[tid + s]; __syncthreads(); }
    float scale = (pool[bc] * (1.f / SP)) / red[0];   // softmax * mean(k)
    unsigned short* yp = yg + (size_t)bc * SP;
#pragma unroll
    for (int t = 0; t < 8; ++t) {
        int col = t * 16 + l15;
#pragma unroll
        for (int e = 0; e < 8; ++e) {
            int row = M0 + e + 8 * half;
            yp[row * WH + col] = f2bf(acc[t][e] * scale);
        }
    }
}

// ---------------- kernel 3: squeeze 1x1 conv (M=32,K=256 WMMA) + bias + per-(b,d) stats
__global__ void squeeze_kernel(const unsigned short* __restrict__ yg,
                               const float* __restrict__ w_sq, const float* __restrict__ b_sq,
                               float* __restrict__ sg, float* __restrict__ stats) {
    __shared__ alignas(32) unsigned short As[CR_N * C_N];   // 16KB, w_sq bf16 row-major [d][c]
    __shared__ alignas(32) unsigned short yts[128 * 128];   // 32KB, y^T tile [n][ck]
    __shared__ float lsum[CR_N], lsq[CR_N];
    int b  = blockIdx.x >> 7;
    int p0 = (blockIdx.x & 127) * 128;
    int tid = threadIdx.x, wave = tid >> 5, lane = tid & 31, half = lane >> 4, l15 = lane & 15;
    for (int i = tid; i < CR_N * C_N; i += 256) As[i] = f2bf(w_sq[i]);
    if (tid < CR_N) { lsum[tid] = 0.f; lsq[tid] = 0.f; }
    int mt  = wave >> 2;
    int nt0 = (wave * 2) & 7;
    v8f acc0, acc1;
#pragma unroll
    for (int e = 0; e < 8; ++e) { acc0[e] = 0.f; acc1[e] = 0.f; }
    for (int kc = 0; kc < 2; ++kc) {
        __syncthreads();
        for (int i = tid; i < 16384; i += 256) {     // stage transposed y tile
            int ck = i >> 7, n = i & 127;
            yts[n * 128 + ck] = yg[(size_t)(b * C_N + kc * 128 + ck) * SP + p0 + n];
        }
        __syncthreads();
#pragma unroll
        for (int kk = 0; kk < 4; ++kk) {
            int k0l = kk * 32;
            int k0g = kc * 128 + k0l;
            v16bf a  = ld16(&As[(mt * 16 + l15) * C_N + k0g + half * 16]);
            v16bf b0 = ld16(&yts[(nt0 * 16 + l15) * 128 + k0l + half * 16]);
            v16bf b1 = ld16(&yts[((nt0 + 1) * 16 + l15) * 128 + k0l + half * 16]);
            acc0 = __builtin_amdgcn_wmma_f32_16x16x32_bf16(false, a, false, b0, (short)0, acc0, false, false);
            acc1 = __builtin_amdgcn_wmma_f32_16x16x32_bf16(false, a, false, b1, (short)0, acc1, false, false);
        }
    }
    __syncthreads();
#pragma unroll
    for (int j = 0; j < 2; ++j) {
        v8f& ac = j ? acc1 : acc0;
        int col = p0 + (nt0 + j) * 16 + l15;
#pragma unroll
        for (int e = 0; e < 8; ++e) {
            int d = mt * 16 + e + 8 * half;
            float v = ac[e] + b_sq[d];
            sg[(size_t)(b * CR_N + d) * SP + col] = v;
            atomicAdd(&lsum[d], v);
            atomicAdd(&lsq[d], v * v);
        }
    }
    __syncthreads();
    if (tid < CR_N) {
        atomicAdd(&stats[b * CR_N + tid],               lsum[tid]);
        atomicAdd(&stats[B_N * CR_N + b * CR_N + tid],  lsq[tid]);
    }
}

// ---------------- kernel 4: InstanceNorm + ReLU + expand 1x1 conv (M=256,K=32 WMMA) + bias
__global__ void expand_kernel(const float* __restrict__ sg, const float* __restrict__ stats,
                              const float* __restrict__ w_ex, const float* __restrict__ b_ex,
                              float* __restrict__ out) {
    __shared__ alignas(32) unsigned short Ae[C_N * CR_N];   // 16KB, w_ex bf16 [c][d]
    __shared__ alignas(32) unsigned short sn[128 * CR_N];   // 8KB, normalized s^T [n][d]
    __shared__ float mu_s[CR_N], rs_s[CR_N];
    int b  = blockIdx.x >> 7;
    int p0 = (blockIdx.x & 127) * 128;
    int tid = threadIdx.x, wave = tid >> 5, lane = tid & 31, half = lane >> 4, l15 = lane & 15;
    for (int i = tid; i < C_N * CR_N; i += 256) Ae[i] = f2bf(w_ex[i]);
    if (tid < CR_N) {
        float su = stats[b * CR_N + tid];
        float sq = stats[B_N * CR_N + b * CR_N + tid];
        float mu = su * (1.f / SP);
        float var = sq * (1.f / SP) - mu * mu;
        mu_s[tid] = mu;
        rs_s[tid] = rsqrtf(var + 1e-5f);
    }
    __syncthreads();
    for (int i = tid; i < 128 * CR_N; i += 256) {
        int d = i >> 7, n = i & 127;
        float v = sg[(size_t)(b * CR_N + d) * SP + p0 + n];
        v = (v - mu_s[d]) * rs_s[d];
        sn[n * CR_N + d] = f2bf(fmaxf(v, 0.f));
    }
    __syncthreads();
    v16bf bm[8];
#pragma unroll
    for (int nt = 0; nt < 8; ++nt) bm[nt] = ld16(&sn[(nt * 16 + l15) * CR_N + half * 16]);
#pragma unroll
    for (int j = 0; j < 2; ++j) {
        int mt = wave * 2 + j;
        v16bf a = ld16(&Ae[(mt * 16 + l15) * CR_N + half * 16]);
#pragma unroll
        for (int nt = 0; nt < 8; ++nt) {
            v8f acc;
#pragma unroll
            for (int e = 0; e < 8; ++e) acc[e] = 0.f;
            acc = __builtin_amdgcn_wmma_f32_16x16x32_bf16(false, a, false, bm[nt], (short)0, acc, false, false);
            int col = p0 + nt * 16 + l15;
#pragma unroll
            for (int e = 0; e < 8; ++e) {
                int c = mt * 16 + e + 8 * half;
                out[(size_t)(b * C_N + c) * SP + col] = acc[e] + b_ex[c];
            }
        }
    }
}

extern "C" void kernel_launch(void* const* d_in, const int* in_sizes, int n_in,
                              void* d_out, int out_size, void* d_ws, size_t ws_size,
                              hipStream_t stream) {
    (void)in_sizes; (void)n_in; (void)out_size; (void)ws_size;
    const float* x    = (const float*)d_in[0];
    const float* wq   = (const float*)d_in[1];
    const float* bq   = (const float*)d_in[2];
    const float* wk   = (const float*)d_in[3];
    const float* bk   = (const float*)d_in[4];
    const float* w_sq = (const float*)d_in[5];
    const float* b_sq = (const float*)d_in[6];
    const float* w_ex = (const float*)d_in[7];
    const float* b_ex = (const float*)d_in[8];
    float* out = (float*)d_out;

    char* ws = (char*)d_ws;
    size_t off = 0;
    unsigned short* qg  = (unsigned short*)(ws + off); off += (size_t)B_N * C_N * SP * 2;   // 64MB
    unsigned short* ktg = (unsigned short*)(ws + off); off += (size_t)B_N * C_N * SP * 2;   // 64MB
    unsigned short* yg  = (unsigned short*)(ws + off); off += (size_t)B_N * C_N * SP * 2;   // 64MB
    float* sg    = (float*)(ws + off); off += (size_t)B_N * CR_N * SP * 4;                  // 16MB
    float* pool  = (float*)(ws + off); off += (size_t)B_N * C_N * 4;
    float* stats = (float*)(ws + off); off += (size_t)2 * B_N * CR_N * 4;

    init_stats_kernel<<<8, 256, 0, stream>>>(pool, stats);
    conv_qk_kernel<<<B_N * C_N * 4, 256, 0, stream>>>(x, wq, bq, wk, bk, qg, ktg, pool);
    qk_gemm_softmax_kernel<<<B_N * C_N, 256, 0, stream>>>(qg, ktg, pool, yg);
    squeeze_kernel<<<B_N * 128, 256, 0, stream>>>(yg, w_sq, b_sq, sg, stats);
    expand_kernel<<<B_N * 128, 256, 0, stream>>>(sg, stats, w_ex, b_ex, out);
}
